// GATConvModel_49031346651833
// MI455X (gfx1250) — matrix operand order repaired and checked by reading, not compiled, not run on hardware
//
#include <hip/hip_runtime.h>
#include <hip/hip_bf16.h>

// ---------------------------------------------------------------------------
// GAT (2-layer, edge-attr, self-loop fill='mean', residual) for MI455X gfx1250
//   - Node GEMMs: v_wmma_f32_16x16x32_bf16, one 16-row strip per wave.
//     A: clamped unguarded b128 loads + inline f32->bf16.
//     B: weights pre-swizzled once into WMMA lane layout (packed bf16),
//        loaded as 2x b128 per fragment -- zero conversion VALU in the loop.
//   - Edge path algebraically folded to 3 scalars/edge (exact linear folding)
//   - Segment softmax: atomicMax on monotonic uint key, then exp-sum, scatter
// ---------------------------------------------------------------------------

typedef __attribute__((ext_vector_type(16))) __bf16 v16bf;
typedef __attribute__((ext_vector_type(8)))  float  v8f;

__device__ __forceinline__ __bf16 f2bf(float f) {
  union { float f; unsigned u; } in; in.f = f;
  unsigned r = (in.u + 0x7FFFu + ((in.u >> 16) & 1u)) >> 16;   // RNE
  union { unsigned short s; __bf16 b; } out; out.s = (unsigned short)r;
  return out.b;
}
__device__ __forceinline__ unsigned fkey(float f) {            // order-preserving
  unsigned u = __float_as_uint(f);
  return (u & 0x80000000u) ? ~u : (u | 0x80000000u);
}
__device__ __forceinline__ float funkey(unsigned k) {
  unsigned u = (k & 0x80000000u) ? (k & 0x7FFFFFFFu) : ~k;
  return __uint_as_float(u);
}
__device__ __forceinline__ float lrelu(float x, float s) { return x > 0.f ? x : s * x; }

// ---------------- weight pre-swizzle into WMMA B-fragment layout ------------
// P[((nt*2+s)*32 + lane)*16 + i] = bf16(W[K*NOUT + col]),
//   col = nt*16 + lane%16,  K = 32*s + 8*(lane/16) + (i%8) + 16*(i/8)   (B 32x16 ISA layout)
__global__ void k_pack_w(const float* __restrict__ W, unsigned short* __restrict__ P,
                         int NOUT) {
  int idx = blockIdx.x * blockDim.x + threadIdx.x;
  if (idx >= 64 * NOUT) return;
  int i    = idx & 15;
  int lane = (idx >> 4) & 31;
  int fs   = idx >> 9;                 // nt*2 + s
  int nt   = fs >> 1, s = fs & 1;
  int col  = nt * 16 + (lane & 15);
  int K    = 32 * s + 8 * (lane >> 4) + (i & 7) + ((i >> 3) << 4);
  union { __bf16 b; unsigned short u; } cv;
  cv.b = f2bf(W[(size_t)K * NOUT + col]);
  P[idx] = cv.u;
}

// ---------------- WMMA GEMM: C[M,NOUT] = A[M,64] @ W[64,NOUT] (+bias) -------
// One wave computes a full 16 x NOUT strip. Tail A-rows are clamped (their
// garbage only lands in D rows we never store), so all loads are unguarded.
template <int NOUT>
__global__ void k_gemm_bf16(const float* __restrict__ A,
                            const unsigned short* __restrict__ P,
                            const float* __restrict__ bias, float* __restrict__ C,
                            int M) {
  constexpr int K = 64;
  constexpr int NT = NOUT / 16;
  int wave = (int)((blockIdx.x * blockDim.x + threadIdx.x) >> 5);
  int mtiles = (M + 15) >> 4;
  if (wave >= mtiles) return;                       // wave-uniform exit
  int m0 = wave << 4;
  int lane = threadIdx.x & 31;
  int lrow = lane & 15, lhi = lane >> 4;

  int arow = min(m0 + lrow, M - 1);                 // clamp: no OOB, no exec mask
  const float* Arow = A + (size_t)arow * K;

  // Both k-step A fragments, loaded once as 4x b128 each.
  v16bf af[2];
#pragma unroll
  for (int s = 0; s < 2; ++s) {
    int ka = 32 * s + 8 * lhi;                      // ISA 16-bit A 16x32 layout
    float4 p0 = *(const float4*)(Arow + ka);
    float4 p1 = *(const float4*)(Arow + ka + 4);
    float4 q0 = *(const float4*)(Arow + ka + 16);
    float4 q1 = *(const float4*)(Arow + ka + 20);
    float t[16] = {p0.x, p0.y, p0.z, p0.w, p1.x, p1.y, p1.z, p1.w,
                   q0.x, q0.y, q0.z, q0.w, q1.x, q1.y, q1.z, q1.w};
#pragma unroll
    for (int i = 0; i < 16; ++i) af[s][i] = f2bf(t[i]);
  }

  v8f acc[NT];
#pragma unroll
  for (int nt = 0; nt < NT; ++nt) acc[nt] = (v8f){};

#pragma unroll
  for (int nt = 0; nt < NT; ++nt) {
#pragma unroll
    for (int s = 0; s < 2; ++s) {
      const uint4* Pf = (const uint4*)(P + ((size_t)(nt * 2 + s) * 32 + lane) * 16);
      union { uint4 q[2]; v16bf v; } u;
      u.q[0] = Pf[0]; u.q[1] = Pf[1];               // 16 bf16 = 2x b128
      acc[nt] = __builtin_amdgcn_wmma_f32_16x16x32_bf16(false, af[s], false, u.v,
                                                        (short)0, acc[nt], false, false);
    }
  }

  bool full = (m0 + 16) <= M;                       // uniform fast path
#pragma unroll
  for (int nt = 0; nt < NT; ++nt) {
    float bv = bias ? bias[nt * 16 + lrow] : 0.f;
    if (full) {
#pragma unroll
      for (int r = 0; r < 8; ++r) {
        int row = m0 + r + 8 * lhi;                 // C/D: VGPR r -> M=r / r+8
        C[(size_t)row * NOUT + nt * 16 + lrow] = acc[nt][r] + bv;
      }
    } else {
#pragma unroll
      for (int r = 0; r < 8; ++r) {
        int row = m0 + r + 8 * lhi;
        if (row < M) C[(size_t)row * NOUT + nt * 16 + lrow] = acc[nt][r] + bv;
      }
    }
  }
}

// ---------------- weight folding (tiny, one block) --------------------------
// fold[0..31]=We@u1, [32..63]=We@u2h0, [64..95]=We@u2h1, [96..98]=be·u{1,2a,2b}
__global__ void k_fold(const float* We, const float* be, const float* We1,
                       const float* ae1, const float* We2, const float* ae2,
                       float* fold) {
  __shared__ float u1[64], u2a[64], u2b[64];
  int t = threadIdx.x;
  if (t < 64) {
    float s1 = 0.f, s2a = 0.f, s2b = 0.f;
    for (int j = 0; j < 64; ++j) {
      s1  += We1[t * 64 + j]        * ae1[j];
      s2a += We2[t * 128 + j]       * ae2[j];
      s2b += We2[t * 128 + 64 + j]  * ae2[64 + j];
    }
    u1[t] = s1; u2a[t] = s2a; u2b[t] = s2b;
  }
  __syncthreads();
  if (t < 32) {
    float w1 = 0.f, w2a = 0.f, w2b = 0.f;
    for (int k = 0; k < 64; ++k) {
      float we = We[t * 64 + k];
      w1 += we * u1[k]; w2a += we * u2a[k]; w2b += we * u2b[k];
    }
    fold[t] = w1; fold[32 + t] = w2a; fold[64 + t] = w2b;
  }
  if (t == 0) {
    float c1 = 0.f, c2a = 0.f, c2b = 0.f;
    for (int k = 0; k < 64; ++k) {
      c1 += be[k] * u1[k]; c2a += be[k] * u2a[k]; c2b += be[k] * u2b[k];
    }
    fold[96] = c1; fold[97] = c2a; fold[98] = c2b;
  }
}

// ---------------- edge scalars + degree / self-loop sums --------------------
__global__ void k_edge_ae(const float* __restrict__ eattr, const int* __restrict__ dst,
                          const float* __restrict__ fold, float* __restrict__ ae1,
                          float* __restrict__ ae2, float* __restrict__ cnt,
                          float* __restrict__ s1, float* __restrict__ s2, int E) {
  int e = blockIdx.x * blockDim.x + threadIdx.x;
  if (e >= E) return;
  const float4* a = (const float4*)(eattr + (size_t)e * 32);
  float d1 = fold[96], d2a = fold[97], d2b = fold[98];
#pragma unroll
  for (int q = 0; q < 8; ++q) {
    float4 v = a[q];
    const float* f1 = fold + 4 * q;
    const float* f2 = fold + 32 + 4 * q;
    const float* f3 = fold + 64 + 4 * q;
    d1  += v.x * f1[0] + v.y * f1[1] + v.z * f1[2] + v.w * f1[3];
    d2a += v.x * f2[0] + v.y * f2[1] + v.z * f2[2] + v.w * f2[3];
    d2b += v.x * f3[0] + v.y * f3[1] + v.z * f3[2] + v.w * f3[3];
  }
  ae1[e] = d1; ae2[2 * e] = d2a; ae2[2 * e + 1] = d2b;
  int d = dst[e];
  atomicAdd(&cnt[d], 1.0f);
  atomicAdd(&s1[d], d1);
  atomicAdd(&s2[2 * d], d2a);
  atomicAdd(&s2[2 * d + 1], d2b);
}

// ---------------- conv1 (heads = 1) -----------------------------------------
__global__ void k_node_prep1(const float* __restrict__ hW, const float* __restrict__ as_,
                             const float* __restrict__ ad_, const float* __restrict__ cnt,
                             const float* __restrict__ s1, float* asrc, float* adst,
                             float* self_, unsigned* mkey, int N) {
  int n = blockIdx.x * blockDim.x + threadIdx.x;
  if (n >= N) return;
  float sa = 0.f, sd = 0.f;
  const float4* h = (const float4*)(hW + (size_t)n * 64);
#pragma unroll
  for (int q = 0; q < 16; ++q) {
    float4 v = h[q];
    const float* ap = as_ + 4 * q;
    const float* dp = ad_ + 4 * q;
    sa += v.x * ap[0] + v.y * ap[1] + v.z * ap[2] + v.w * ap[3];
    sd += v.x * dp[0] + v.y * dp[1] + v.z * dp[2] + v.w * dp[3];
  }
  asrc[n] = sa; adst[n] = sd;
  float loop = s1[n] / fmaxf(cnt[n], 1.0f);
  float l = lrelu(sa + sd + loop, 0.2f);
  self_[n] = l; mkey[n] = fkey(l);
}

__global__ void k_edge_max1(const int* src, const int* dst, const float* asrc,
                            const float* adst, const float* ae1, float* logit,
                            unsigned* mkey, int E) {
  int e = blockIdx.x * blockDim.x + threadIdx.x;
  if (e >= E) return;
  int s = src[e], d = dst[e];
  float l = lrelu(asrc[s] + adst[d] + ae1[e], 0.2f);
  logit[e] = l;
  atomicMax(&mkey[d], fkey(l));
}

__global__ void k_node_denom1(const unsigned* mkey, const float* self_, float* m,
                              float* denom, int N) {
  int n = blockIdx.x * blockDim.x + threadIdx.x;
  if (n >= N) return;
  float mv = funkey(mkey[n]);
  m[n] = mv; denom[n] = __expf(self_[n] - mv);
}

__global__ void k_edge_denom1(const int* dst, const float* logit, const float* m,
                              float* denom, int E) {
  int e = blockIdx.x * blockDim.x + threadIdx.x;
  if (e >= E) return;
  int d = dst[e];
  atomicAdd(&denom[d], __expf(logit[e] - m[d]));
}

__global__ void k_edge_alpha1(const int* dst, const float* m, const float* denom,
                              float* logit, int E) {
  int e = blockIdx.x * blockDim.x + threadIdx.x;
  if (e >= E) return;
  int d = dst[e];
  logit[e] = __expf(logit[e] - m[d]) / (denom[d] + 1e-16f);
}

__global__ void k_out_init1(const float* self_, const float* m, const float* denom,
                            const float* hW, const float* hR, float* out, int N) {
  long long idx = (long long)blockIdx.x * blockDim.x + threadIdx.x;
  if (idx >= (long long)N * 64) return;
  int n = (int)(idx >> 6);
  float a = __expf(self_[n] - m[n]) / (denom[n] + 1e-16f);
  out[idx] = a * hW[idx] + hR[idx];
}

__global__ void k_scatter1(const int* src, const int* dst, const float* alpha,
                           const float* hW, float* out, int E) {
  long long idx = (long long)blockIdx.x * blockDim.x + threadIdx.x;
  if (idx >= (long long)E * 64) return;
  int e = (int)(idx >> 6), j = (int)(idx & 63);
  int s = src[e], d = dst[e];
  atomicAdd(&out[(size_t)d * 64 + j], alpha[e] * hW[(size_t)s * 64 + j]);
}

__global__ void k_lrelu(float* p, long long n, float slope) {
  long long i = (long long)blockIdx.x * blockDim.x + threadIdx.x;
  if (i >= n) return;
  float v = p[i];
  p[i] = v > 0.f ? v : slope * v;
}

// ---------------- conv2 (heads = 2) -----------------------------------------
__global__ void k_node_prep2(const float* __restrict__ hW, const float* __restrict__ as_,
                             const float* __restrict__ ad_, const float* __restrict__ cnt,
                             const float* __restrict__ s2, float* asrc, float* adst,
                             float* self_, unsigned* mkey, int N) {
  int n = blockIdx.x * blockDim.x + threadIdx.x;
  if (n >= N) return;
  float c = fmaxf(cnt[n], 1.0f);
#pragma unroll
  for (int h = 0; h < 2; ++h) {
    float sa = 0.f, sd = 0.f;
    const float4* hp = (const float4*)(hW + (size_t)n * 128 + h * 64);
#pragma unroll
    for (int q = 0; q < 16; ++q) {
      float4 v = hp[q];
      const float* ap = as_ + h * 64 + 4 * q;
      const float* dp = ad_ + h * 64 + 4 * q;
      sa += v.x * ap[0] + v.y * ap[1] + v.z * ap[2] + v.w * ap[3];
      sd += v.x * dp[0] + v.y * dp[1] + v.z * dp[2] + v.w * dp[3];
    }
    int nh = 2 * n + h;
    asrc[nh] = sa; adst[nh] = sd;
    float l = lrelu(sa + sd + s2[nh] / c, 0.2f);
    self_[nh] = l; mkey[nh] = fkey(l);
  }
}

__global__ void k_edge_max2(const int* src, const int* dst, const float* asrc,
                            const float* adst, const float* ae2, float* logit,
                            unsigned* mkey, int E) {
  int e = blockIdx.x * blockDim.x + threadIdx.x;
  if (e >= E) return;
  int s = src[e], d = dst[e];
#pragma unroll
  for (int h = 0; h < 2; ++h) {
    float l = lrelu(asrc[2 * s + h] + adst[2 * d + h] + ae2[2 * e + h], 0.2f);
    logit[2 * e + h] = l;
    atomicMax(&mkey[2 * d + h], fkey(l));
  }
}

__global__ void k_edge_denom2(const int* dst, const float* logit, const float* m,
                              float* denom, int E) {
  long long idx = (long long)blockIdx.x * blockDim.x + threadIdx.x;
  if (idx >= (long long)E * 2) return;
  int e = (int)(idx >> 1), h = (int)(idx & 1);
  int d = dst[e];
  atomicAdd(&denom[2 * d + h], __expf(logit[idx] - m[2 * d + h]));
}

__global__ void k_edge_alpha2(const int* dst, const float* m, const float* denom,
                              float* logit, int E) {
  long long idx = (long long)blockIdx.x * blockDim.x + threadIdx.x;
  if (idx >= (long long)E * 2) return;
  int e = (int)(idx >> 1), h = (int)(idx & 1);
  int d = dst[e];
  logit[idx] = __expf(logit[idx] - m[2 * d + h]) / (denom[2 * d + h] + 1e-16f);
}

__global__ void k_out_init2(const float* self_, const float* m, const float* denom,
                            const float* hW, const float* hR, float* out, int N) {
  long long idx = (long long)blockIdx.x * blockDim.x + threadIdx.x;
  if (idx >= (long long)N * 128) return;
  int nh = (int)(idx >> 6);   // = 2*n + h
  float a = __expf(self_[nh] - m[nh]) / (denom[nh] + 1e-16f);
  out[idx] = a * hW[idx] + hR[idx];
}

__global__ void k_scatter2(const int* src, const int* dst, const float* alpha,
                           const float* hW, float* out, int E) {
  long long idx = (long long)blockIdx.x * blockDim.x + threadIdx.x;
  if (idx >= (long long)E * 128) return;
  int e = (int)(idx >> 7);
  int hj = (int)(idx & 127);
  int h = hj >> 6;
  int s = src[e], d = dst[e];
  atomicAdd(&out[(size_t)d * 128 + hj], alpha[2 * e + h] * hW[(size_t)s * 128 + hj]);
}

// ---------------- output head ------------------------------------------------
__global__ void k_final(const float* __restrict__ h, const float* __restrict__ Wl,
                        const float* __restrict__ bl, float* __restrict__ y, int N) {
  int n = blockIdx.x * blockDim.x + threadIdx.x;
  if (n >= N) return;
  float acc = bl[0];
  const float4* hp = (const float4*)(h + (size_t)n * 128);
#pragma unroll
  for (int q = 0; q < 32; ++q) {
    float4 v = hp[q];
    const float* w = Wl + 4 * q;
    acc += v.x * w[0] + v.y * w[1] + v.z * w[2] + v.w * w[3];
  }
  y[n] = acc;
}

// ---------------------------------------------------------------------------
extern "C" void kernel_launch(void* const* d_in, const int* in_sizes, int n_in,
                              void* d_out, int out_size, void* d_ws, size_t ws_size,
                              hipStream_t stream) {
  const int N = in_sizes[0] / 64;      // NODE_IN = 64
  const int E = in_sizes[1] / 2;

  const float* x     = (const float*)d_in[0];
  const int*   ei    = (const int*)  d_in[1];
  const int*   src   = ei;
  const int*   dst   = ei + E;
  const float* eattr = (const float*)d_in[2];
  const float* Wn  = (const float*)d_in[3];  const float* bn  = (const float*)d_in[4];
  const float* We  = (const float*)d_in[5];  const float* be  = (const float*)d_in[6];
  const float* W1  = (const float*)d_in[7];  const float* as1 = (const float*)d_in[8];
  const float* ad1 = (const float*)d_in[9];  const float* We1 = (const float*)d_in[10];
  const float* aE1 = (const float*)d_in[11]; const float* Wr1 = (const float*)d_in[12];
  const float* b1  = (const float*)d_in[13];
  const float* W2  = (const float*)d_in[14]; const float* as2 = (const float*)d_in[15];
  const float* ad2 = (const float*)d_in[16]; const float* We2 = (const float*)d_in[17];
  const float* aE2 = (const float*)d_in[18]; const float* Wr2 = (const float*)d_in[19];
  const float* b2  = (const float*)d_in[20];
  const float* Wl  = (const float*)d_in[21]; const float* bl  = (const float*)d_in[22];
  float* y = (float*)d_out;

  // workspace bump allocator (floats), 128B-aligned chunks
  float* ws = (float*)d_ws;
  size_t off = 0;
  auto alloc = [&](size_t nf) { size_t o = off; off += (nf + 31) & ~(size_t)31; return ws + o; };

  float* fold   = alloc(128);
  float* cnt    = alloc((size_t)4 * N);     // cnt | s1 | s2(2N) contiguous -> one memset
  float* s1b    = cnt + N;
  float* s2b    = cnt + 2 * (size_t)N;
  float* h0     = alloc((size_t)N * 64);
  float* hW1    = alloc((size_t)N * 64);
  float* hR1    = alloc((size_t)N * 64);
  float* out1   = alloc((size_t)N * 64);    // becomes h1 after leaky-relu
  float* hW2    = alloc((size_t)N * 128);
  float* hR2    = alloc((size_t)N * 128);
  float* out2   = alloc((size_t)N * 128);
  float* asrc1  = alloc(N);  float* adst1 = alloc(N);
  float* self1  = alloc(N);  float* m1    = alloc(N);  float* den1 = alloc(N);
  unsigned* mk1 = (unsigned*)alloc(N);
  float* asrc2  = alloc(2 * (size_t)N); float* adst2 = alloc(2 * (size_t)N);
  float* self2  = alloc(2 * (size_t)N); float* m2    = alloc(2 * (size_t)N);
  float* den2   = alloc(2 * (size_t)N);
  unsigned* mk2 = (unsigned*)alloc(2 * (size_t)N);
  float* ae1b   = alloc(E);
  float* ae2b   = alloc(2 * (size_t)E);
  float* lg1    = alloc(E);                 // logit1 -> alpha1 in place
  float* lg2    = alloc(2 * (size_t)E);     // logit2 -> alpha2 in place
  unsigned short* pWn  = (unsigned short*)alloc(2048);   // 64x64 bf16
  unsigned short* pW1  = (unsigned short*)alloc(2048);
  unsigned short* pWr1 = (unsigned short*)alloc(2048);
  unsigned short* pW2  = (unsigned short*)alloc(4096);   // 64x128 bf16
  unsigned short* pWr2 = (unsigned short*)alloc(4096);

  const int BT = 256;
  auto blk = [&](long long n) { return (unsigned)((n + BT - 1) / BT); };
  auto gemm64 = [&](const float* A, const unsigned short* P, const float* bias,
                    float* C, int M) {
    long long waves = (M + 15) / 16;
    k_gemm_bf16<64><<<blk(waves * 32), BT, 0, stream>>>(A, P, bias, C, M);
  };
  auto gemm128 = [&](const float* A, const unsigned short* P, const float* bias,
                     float* C, int M) {
    long long waves = (M + 15) / 16;
    k_gemm_bf16<128><<<blk(waves * 32), BT, 0, stream>>>(A, P, bias, C, M);
  };

  // 0) fold edge-attention weights; pre-swizzle GEMM weights; zero accumulators
  k_fold<<<1, 64, 0, stream>>>(We, be, We1, aE1, We2, aE2, fold);
  k_pack_w<<<blk(64 * 64), BT, 0, stream>>>(Wn,  pWn,  64);
  k_pack_w<<<blk(64 * 64), BT, 0, stream>>>(W1,  pW1,  64);
  k_pack_w<<<blk(64 * 64), BT, 0, stream>>>(Wr1, pWr1, 64);
  k_pack_w<<<blk(64 * 128), BT, 0, stream>>>(W2,  pW2,  128);
  k_pack_w<<<blk(64 * 128), BT, 0, stream>>>(Wr2, pWr2, 128);
  hipMemsetAsync(cnt, 0, (size_t)4 * N * sizeof(float), stream);

  // 1) per-edge attention scalars + in-degree + self-loop sums (reads edge_attr once)
  k_edge_ae<<<blk(E), BT, 0, stream>>>(eattr, dst, fold, ae1b, ae2b, cnt, s1b, s2b, E);

  // 2) node encoder (WMMA)
  gemm64(x, pWn, bn, h0, N);

  // 3) conv1
  gemm64(h0, pW1, nullptr, hW1, N);
  gemm64(h0, pWr1, b1, hR1, N);
  k_node_prep1<<<blk(N), BT, 0, stream>>>(hW1, as1, ad1, cnt, s1b, asrc1, adst1, self1, mk1, N);
  k_edge_max1<<<blk(E), BT, 0, stream>>>(src, dst, asrc1, adst1, ae1b, lg1, mk1, E);
  k_node_denom1<<<blk(N), BT, 0, stream>>>(mk1, self1, m1, den1, N);
  k_edge_denom1<<<blk(E), BT, 0, stream>>>(dst, lg1, m1, den1, E);
  k_edge_alpha1<<<blk(E), BT, 0, stream>>>(dst, m1, den1, lg1, E);
  k_out_init1<<<blk((long long)N * 64), BT, 0, stream>>>(self1, m1, den1, hW1, hR1, out1, N);
  k_scatter1<<<blk((long long)E * 64), BT, 0, stream>>>(src, dst, lg1, hW1, out1, E);
  k_lrelu<<<blk((long long)N * 64), BT, 0, stream>>>(out1, (long long)N * 64, 0.01f);

  // 4) conv2 (heads = 2)
  gemm128(out1, pW2, nullptr, hW2, N);
  gemm128(out1, pWr2, b2, hR2, N);
  k_node_prep2<<<blk(N), BT, 0, stream>>>(hW2, as2, ad2, cnt, s2b, asrc2, adst2, self2, mk2, N);
  k_edge_max2<<<blk(E), BT, 0, stream>>>(src, dst, asrc2, adst2, ae2b, lg2, mk2, E);
  k_node_denom1<<<blk(2 * (long long)N), BT, 0, stream>>>(mk2, self2, m2, den2, 2 * N);
  k_edge_denom2<<<blk((long long)E * 2), BT, 0, stream>>>(dst, lg2, m2, den2, E);
  k_edge_alpha2<<<blk((long long)E * 2), BT, 0, stream>>>(dst, m2, den2, lg2, E);
  k_out_init2<<<blk((long long)N * 128), BT, 0, stream>>>(self2, m2, den2, hW2, hR2, out2, N);
  k_scatter2<<<blk((long long)E * 128), BT, 0, stream>>>(src, dst, lg2, hW2, out2, E);

  // 5) output head
  k_final<<<blk(N), BT, 0, stream>>>(out2, Wl, bl, y, N);
}
